// LSTM_4518305595487
// MI455X (gfx1250) — matrix operand order repaired and checked by reading, not compiled, not run on hardware
//
#include <hip/hip_runtime.h>
#include <hip/hip_bf16.h>
#include <stdint.h>

#define B_   32
#define H_   512
#define E_   512
#define V_   8192
#define KCAT 1536   /* (H+E) x-part + H h-part, concatenated K */
#define XROW 2048   /* per-row layout: ctx[512] | emb[512] | h_slot0[512] | h_slot1[512] */

typedef __attribute__((ext_vector_type(8)))  __bf16 v8bf;
typedef __attribute__((ext_vector_type(16))) __bf16 v16bf;
typedef __attribute__((ext_vector_type(8)))  float  v8f;

// fp32 -> bf16, round-to-nearest-even (pure bit math: guaranteed to compile)
__device__ __forceinline__ __bf16 f2bf(float f) {
    unsigned u = __float_as_uint(f);
    unsigned r = (u + 0x7FFFu + ((u >> 16) & 1u)) >> 16;
    unsigned short s = (unsigned short)r;
    return __builtin_bit_cast(__bf16, s);
}

// CDNA5 has V_TANH_F32 (transcendental unit). Use it when the builtin exists;
// otherwise fall back to v_rcp_f32-based forms (no IEEE divide chain either way).
#if __has_builtin(__builtin_amdgcn_tanhf)
__device__ __forceinline__ float fast_tanh(float x) { return __builtin_amdgcn_tanhf(x); }
__device__ __forceinline__ float fast_sigm(float x) {
    return 0.5f + 0.5f * __builtin_amdgcn_tanhf(0.5f * x);
}
#else
__device__ __forceinline__ float fast_sigm(float x) {
    return __builtin_amdgcn_rcpf(1.0f + __expf(-x));
}
__device__ __forceinline__ float fast_tanh(float x) {
    return 2.0f * __builtin_amdgcn_rcpf(1.0f + __expf(-2.0f * x)) - 1.0f;
}
#endif

// A-fragment (16x32 bf16, MxK) per ISA layout:
// lanes 0-15 : row M=lane,    K chunks {kb+0..7,  kb+16..23}
// lanes 16-31: row M=lane-16, K chunks {kb+8..15, kb+24..31}
// caller passes p = rowptr + kb + lhalf*8 ; we read p[0..7] and p[16..23]
__device__ __forceinline__ v16bf load_a_frag(const __bf16* p) {
    v8bf lo = *(const v8bf*)(p);
    v8bf hi = *(const v8bf*)(p + 16);
    v16bf a;
#pragma unroll
    for (int i = 0; i < 8; ++i) { a[i] = lo[i]; a[i + 8] = hi[i]; }
    return a;
}

// ---------------------------------------------------------------------------
// Phase 0a: one-time weight prep: W_cat = [W_ih | W_hh] -> bf16, W_out -> bf16,
//           bias = b_ih + b_hh
// ---------------------------------------------------------------------------
__global__ void prep_kernel(const float* __restrict__ Wih, const float* __restrict__ Whh,
                            const float* __restrict__ Wo,
                            const float* __restrict__ bih, const float* __restrict__ bhh,
                            __bf16* __restrict__ Wcatb, __bf16* __restrict__ Woutb,
                            float* __restrict__ bias)
{
    const int NW1 = 4 * H_ * KCAT;   // 2048*1536
    const int NW2 = V_ * H_;         // 8192*512
    const int TOT = NW1 + NW2 + 4 * H_;
    for (int idx = blockIdx.x * blockDim.x + threadIdx.x; idx < TOT;
         idx += gridDim.x * blockDim.x) {
        if (idx < NW1) {
            int o = idx / KCAT, k = idx - o * KCAT;
            float v = (k < (H_ + E_)) ? Wih[(size_t)o * (H_ + E_) + k]
                                      : Whh[(size_t)o * H_ + (k - (H_ + E_))];
            Wcatb[idx] = f2bf(v);
        } else if (idx < NW1 + NW2) {
            int j = idx - NW1;
            Woutb[j] = f2bf(Wo[j]);
        } else {
            int o = idx - NW1 - NW2;
            bias[o] = bih[o] + bhh[o];
        }
    }
}

// ---------------------------------------------------------------------------
// Phase 0b: state init: xh rows = [bf16(ctx) | bf16(emb[START=1]) | bf16(ctx) | --],
//           c0 = ctx (fp32), argmax keys = 0, d_out[:,0,:] = 0
// ---------------------------------------------------------------------------
__global__ void init_kernel(const float* __restrict__ ctx, const float* __restrict__ emb,
                            float* __restrict__ cst, __bf16* __restrict__ xh,
                            unsigned long long* __restrict__ argbuf,
                            float* __restrict__ out, int T)
{
    const int N1 = B_ * H_;      // ctx -> ctx-slot, h0-slot, c0
    const int N2 = B_ * E_;      // start-token embedding
    const int N3 = B_ * V_;      // zero logits at t=0
    const int TOT = N1 + N2 + N3 + B_;
    for (int idx = blockIdx.x * blockDim.x + threadIdx.x; idx < TOT;
         idx += gridDim.x * blockDim.x) {
        if (idx < N1) {
            int b = idx >> 9, e = idx & 511;
            float v = ctx[idx];
            __bf16 bv = f2bf(v);
            xh[(size_t)b * XROW + e]        = bv;   // context slot
            xh[(size_t)b * XROW + 1024 + e] = bv;   // h0 -> slot 0
            cst[idx] = v;                           // c0
        } else if (idx < N1 + N2) {
            int j = idx - N1; int b = j >> 9, e = j & 511;
            xh[(size_t)b * XROW + 512 + e] = f2bf(emb[(size_t)1 * E_ + e]);
        } else if (idx < N1 + N2 + N3) {
            int j = idx - N1 - N2; int b = j >> 13, v = j & (V_ - 1);
            out[(size_t)b * T * V_ + v] = 0.0f;     // out[b, 0, v]
        } else {
            argbuf[idx - N1 - N2 - N3] = 0ULL;
        }
    }
}

// ---------------------------------------------------------------------------
// K1: fused recurrence. grid = H/16 blocks; 8 waves = (gate i/f/g/o) x (batch half).
// gates[32,2048] = [x|h_{t-1}] @ W_cat^T via v_wmma_f32_16x16x32_bf16 (K=1536),
// then i/f/g/o activations + cell update; writes c (fp32) and h_t (bf16, slot hout).
// Block 0 also resets the argmax keys for the *upcoming* K2 (safe: K3 of the
// previous step, which consumed them, ran before this launch).
// ---------------------------------------------------------------------------
__global__ void __launch_bounds__(256)
lstm_step_kernel(const __bf16* __restrict__ Wcatb, __bf16* __restrict__ xh,
                 float* __restrict__ cst, const float* __restrict__ bias,
                 unsigned long long* __restrict__ argbuf,
                 int hin_off, int hout_off)
{
    __shared__ float gLds[4][B_][16];

    if (blockIdx.x == 0 && threadIdx.x < B_) argbuf[threadIdx.x] = 0ULL;

    const int lane  = threadIdx.x & 31;
    const int wave  = threadIdx.x >> 5;
    const int g     = wave >> 1;          // 0..3 : i,f,g,o
    const int mt    = wave & 1;           // batch half
    const int lhalf = lane >> 4;
    const int lmod  = lane & 15;
    const int j0    = blockIdx.x * 16;    // hidden-unit tile
    const int o0    = g * H_ + j0;        // gate-block column base

    const __bf16* arow = xh    + (size_t)(mt * 16 + lmod) * XROW;
    const __bf16* brow = Wcatb + (size_t)(o0 + lmod) * KCAT;

    // two accumulators break the WMMA SRC2 dependency chain (even/odd K tiles)
    v8f acc0 = {}, acc1 = {};
#pragma unroll 4
    for (int kb = 0; kb < KCAT; kb += 64) {
        // K 0..1023 reads ctx|emb; K 1024.. reads the double-buffered h slot
        const int ab0 = (kb < (H_ + E_)) ? kb : (hin_off + (kb - (H_ + E_)));
        const int kb1 = kb + 32;
        const int ab1 = (kb1 < (H_ + E_)) ? kb1 : (hin_off + (kb1 - (H_ + E_)));
        v16bf a0 = load_a_frag(arow + ab0 + lhalf * 8);
        v16bf b0 = load_a_frag(brow + kb  + lhalf * 8);
        v16bf a1 = load_a_frag(arow + ab1 + lhalf * 8);
        v16bf b1 = load_a_frag(brow + kb1 + lhalf * 8);
        __builtin_prefetch(brow + kb + 256, 0, 3);   // global_prefetch_b8 (near)
        acc0 = __builtin_amdgcn_wmma_f32_16x16x32_bf16(false, a0, false, b0,
                                                       (short)0, acc0, false, false);
        acc1 = __builtin_amdgcn_wmma_f32_16x16x32_bf16(false, a1, false, b1,
                                                       (short)0, acc1, false, false);
    }
    v8f acc = acc0 + acc1;

    // C/D layout: VGPR r -> (lanes 0-15: M=r, N=lane) (lanes 16-31: M=r+8, N=lane-16)
#pragma unroll
    for (int r = 0; r < 8; ++r)
        gLds[g][mt * 16 + lhalf * 8 + r][lmod] = acc[r];
    __syncthreads();

    for (int idx = threadIdx.x; idx < B_ * 16; idx += 256) {
        const int m = idx >> 4, col = idx & 15;
        const int j = j0 + col;
        const float gi = gLds[0][m][col] + bias[0 * H_ + j];
        const float gf = gLds[1][m][col] + bias[1 * H_ + j];
        const float gg = gLds[2][m][col] + bias[2 * H_ + j];
        const float go = gLds[3][m][col] + bias[3 * H_ + j];
        const float cold = cst[m * H_ + j];
        const float cn = fast_sigm(gf) * cold + fast_sigm(gi) * fast_tanh(gg);
        const float hn = fast_sigm(go) * fast_tanh(cn);
        cst[m * H_ + j] = cn;
        xh[(size_t)m * XROW + hout_off + j] = f2bf(hn);
    }
}

// ---------------------------------------------------------------------------
// K2: logits[32,8192] = h_t @ W_out^T + b_out. 128 blocks x 64 columns.
// Writes fp32 logits to d_out[:,t,:] and folds the argmax: monotonic f32 key
// packed with ~col (first-index tie-break, matching jnp.argmax), reduced via
// LDS u64 atomicMax then one global u64 atomicMax per batch row.
// ---------------------------------------------------------------------------
__global__ void __launch_bounds__(256)
logits_kernel(const __bf16* __restrict__ Woutb, const __bf16* __restrict__ xh,
              const float* __restrict__ bout, float* __restrict__ out,
              unsigned long long* __restrict__ argbuf, int t, int h_off, int T)
{
    __shared__ unsigned long long sKey[B_];
    if (threadIdx.x < B_) sKey[threadIdx.x] = 0ULL;
    __syncthreads();

    const int lane  = threadIdx.x & 31;
    const int wave  = threadIdx.x >> 5;
    const int nt    = wave >> 1;          // 0..3 column sub-tile
    const int mt    = wave & 1;           // batch half
    const int lhalf = lane >> 4;
    const int lmod  = lane & 15;
    const int v0    = blockIdx.x * 64 + nt * 16;

    const __bf16* arow = xh    + (size_t)(mt * 16 + lmod) * XROW + h_off;
    const __bf16* brow = Woutb + (size_t)(v0 + lmod) * H_;

    v8f acc0 = {}, acc1 = {};
#pragma unroll 4
    for (int kb = 0; kb < H_; kb += 64) {
        v16bf a0 = load_a_frag(arow + kb + lhalf * 8);
        v16bf b0 = load_a_frag(brow + kb + lhalf * 8);
        v16bf a1 = load_a_frag(arow + kb + 32 + lhalf * 8);
        v16bf b1 = load_a_frag(brow + kb + 32 + lhalf * 8);
        acc0 = __builtin_amdgcn_wmma_f32_16x16x32_bf16(false, a0, false, b0,
                                                       (short)0, acc0, false, false);
        acc1 = __builtin_amdgcn_wmma_f32_16x16x32_bf16(false, a1, false, b1,
                                                       (short)0, acc1, false, false);
    }
    v8f acc = acc0 + acc1;

    const int col = v0 + lmod;
    const float bb = bout[col];
    const unsigned long long colkey = (unsigned long long)(unsigned)(~col);
#pragma unroll
    for (int r = 0; r < 8; ++r) {
        const int m = mt * 16 + lhalf * 8 + r;
        const float val = acc[r] + bb;
        out[(size_t)m * T * V_ + (size_t)t * V_ + col] = val;
        unsigned u = __float_as_uint(val);
        u = (u & 0x80000000u) ? ~u : (u | 0x80000000u);   // order-preserving map
        atomicMax(&sKey[m], ((unsigned long long)u << 32) | colkey);
    }
    __syncthreads();
    if (threadIdx.x < B_) atomicMax(&argbuf[threadIdx.x], sKey[threadIdx.x]);
}

// ---------------------------------------------------------------------------
// K3: greedy feedback. One block per batch row: decode that row's winner and
// gather its embedding row (fp32->bf16) into the feedback slot. The argmax
// buffer is reset by the NEXT step's K1 (no cross-block hazard here).
// ---------------------------------------------------------------------------
__global__ void __launch_bounds__(256)
gather_kernel(const float* __restrict__ emb, __bf16* __restrict__ xh,
              const unsigned long long* __restrict__ argbuf)
{
    __shared__ int sIdx;
    const int b = blockIdx.x;
    if (threadIdx.x == 0)
        sIdx = (int)(~(unsigned)(argbuf[b] & 0xFFFFFFFFull));
    __syncthreads();
    const float* src = emb + (size_t)sIdx * E_;
    __bf16* dst = xh + (size_t)b * XROW + 512;
#pragma unroll
    for (int e = threadIdx.x; e < E_; e += 256)
        dst[e] = f2bf(src[e]);
}

// ---------------------------------------------------------------------------
extern "C" void kernel_launch(void* const* d_in, const int* in_sizes, int n_in,
                              void* d_out, int out_size, void* d_ws, size_t ws_size,
                              hipStream_t stream)
{
    (void)in_sizes; (void)n_in; (void)ws_size;
    const float* ctx  = (const float*)d_in[0];   // [32,512]
    const float* embt = (const float*)d_in[1];   // [8192,512]
    const float* Wih  = (const float*)d_in[2];   // [2048,1024]
    const float* Whh  = (const float*)d_in[3];   // [2048,512]
    const float* bih  = (const float*)d_in[4];   // [2048]
    const float* bhh  = (const float*)d_in[5];   // [2048]
    const float* Wo   = (const float*)d_in[6];   // [8192,512]
    const float* bo   = (const float*)d_in[7];   // [8192]
    float* out = (float*)d_out;                  // [32,T,8192]

    const int T = out_size / (B_ * V_);          // device scalar unreadable under capture

    char* ws = (char*)d_ws;
    size_t off = 0;
    __bf16* Wcatb = (__bf16*)(ws + off); off += (size_t)4 * H_ * KCAT * 2;  // 6.29 MB
    __bf16* Woutb = (__bf16*)(ws + off); off += (size_t)V_ * H_ * 2;        // 8.39 MB
    __bf16* xh    = (__bf16*)(ws + off); off += (size_t)B_ * XROW * 2;      // 128 KB
    float*  cst   = (float*) (ws + off); off += (size_t)B_ * H_ * 4;        // 64 KB
    float*  bias  = (float*) (ws + off); off += (size_t)4 * H_ * 4;         // 8 KB
    unsigned long long* argbuf = (unsigned long long*)(ws + off); off += B_ * 8;

    prep_kernel<<<7168, 256, 0, stream>>>(Wih, Whh, Wo, bih, bhh, Wcatb, Woutb, bias);
    init_kernel<<<1160, 256, 0, stream>>>(ctx, embt, cst, xh, argbuf, out, T);

    for (int t = 1; t < T; ++t) {
        const int hin  = 1024 + ((t - 1) & 1) * 512;   // h double-buffer slots
        const int hout = 1024 + (t & 1) * 512;
        lstm_step_kernel<<<H_ / 16, 256, 0, stream>>>(Wcatb, xh, cst, bias, argbuf,
                                                      hin, hout);
        logits_kernel<<<V_ / 64, 256, 0, stream>>>(Woutb, xh, bo, out, argbuf, t, hout, T);
        if (t + 1 < T)
            gather_kernel<<<B_, 256, 0, stream>>>(embt, xh, argbuf);
    }
}
